// Engram_42915313221887
// MI455X (gfx1250) — matrix operand order repaired and checked by reading, not compiled, not run on hardware
//
#include <hip/hip_runtime.h>
#include <hip/hip_bf16.h>
#include <math.h>

// ---------------------------------------------------------------------------
// Engram block for MI455X (gfx1250, wave32, WMMA bf16)
//   tokens BT = B*T = 16384, HIDDEN = 2048, E-feat = 512, heads = 8 x 64
// Kernel 1: one-shot Wk/Wv f32 -> bf16 conversion into d_ws (4 MiB).
// Kernel 2: fused hash+gather -> key GEMM (reduced on the fly) -> gate ->
//           value GEMM -> gated store. One workgroup = 32 tokens, 8 waves.
// ---------------------------------------------------------------------------

typedef __attribute__((ext_vector_type(16))) __bf16 v16bf;
typedef __attribute__((ext_vector_type(8)))  __bf16 v8bf;
typedef __attribute__((ext_vector_type(8)))  float  v8f;

#define HID      2048
#define EFEAT    512
#define PER_HEAD 64
#define M_BLK    32
#define WAVES    8
#define NTILES   (HID / 16)     // 128 d-tiles
#define KSTEPS   (EFEAT / 32)   // 16 k-steps of 32
#define A_STRIDE (EFEAT + 8)    // +16B pad per row -> conflict-free ds_read_b128
#define RMS_EPS  1.1920929e-07f
#define W_ELEMS  (HID * EFEAT)  // 1M elements per weight matrix

__device__ __forceinline__ __bf16 f2bf(float x) {
  // round-to-nearest-even f32 -> bf16
  unsigned u = __builtin_bit_cast(unsigned, x);
  unsigned r = (u + 0x7FFFu + ((u >> 16) & 1u)) >> 16;
  unsigned short h = (unsigned short)r;
  return __builtin_bit_cast(__bf16, h);
}

// ---- one-shot weight conversion: Wk,Wv f32 -> bf16 in workspace ----------
extern "C" __global__ __launch_bounds__(256)
void convert_weights(const float* __restrict__ Wk,
                     const float* __restrict__ Wv,
                     __bf16* __restrict__ Wkb,
                     __bf16* __restrict__ Wvb)
{
  int i = (blockIdx.x * 256 + threadIdx.x) * 4;   // 4 elems / thread
  if (i < W_ELEMS) {
    float4 a = *(const float4*)(Wk + i);
    float4 b = *(const float4*)(Wv + i);
    v8bf pk;
    pk[0]=f2bf(a.x); pk[1]=f2bf(a.y); pk[2]=f2bf(a.z); pk[3]=f2bf(a.w);
    pk[4]=0; pk[5]=0; pk[6]=0; pk[7]=0;
    *(uint2*)(Wkb + i) = *(uint2*)&pk;            // 8B store of 4 bf16
    v8bf pv;
    pv[0]=f2bf(b.x); pv[1]=f2bf(b.y); pv[2]=f2bf(b.z); pv[3]=f2bf(b.w);
    pv[4]=0; pv[5]=0; pv[6]=0; pv[7]=0;
    *(uint2*)(Wvb + i) = *(uint2*)&pv;
  }
}

// A fragment (16x32 bf16, ISA 7.12.2): lane l holds row (l&15);
// half kh=l>>4: elems 0..15 = K kh*16+0..15 (contiguous, matches B).
__device__ __forceinline__ v16bf load_afrag(const __bf16* A, int rowbase, int f0, int lane) {
  int row = rowbase + (lane & 15);
  int kh  = (lane >> 4) * 16;
  const __bf16* p = A + row * A_STRIDE + f0;
  v8bf lo = *(const v8bf*)(p + kh);        // 16B-aligned ds_read_b128
  v8bf hi = *(const v8bf*)(p + kh + 8);
  return __builtin_shufflevector(lo, hi, 0,1,2,3,4,5,6,7,8,9,10,11,12,13,14,15);
}

// B fragment (32x16 bf16): lane l = column d0+(l&15); 16 contiguous K
// values (l>>4)*16 + 0..15 -> one 32B contiguous bf16 load (2x b128).
__device__ __forceinline__ v16bf load_bfrag(const __bf16* __restrict__ W,
                                            int d0, int f0, int lane) {
  const __bf16* p = W + (size_t)(d0 + (lane & 15)) * EFEAT + f0 + (lane >> 4) * 16;
  return *(const v16bf*)p;                 // 32B aligned
}

extern "C" __global__ __launch_bounds__(256)
void engram_fused(const float* __restrict__ hidden,
                  const long long* __restrict__ ids,
                  const float* __restrict__ emb,
                  const __bf16* __restrict__ Wkb,
                  const __bf16* __restrict__ Wvb,
                  const float* __restrict__ kw,
                  const float* __restrict__ qw,
                  const long long* __restrict__ offs,
                  const long long* __restrict__ mults,
                  const long long* __restrict__ mods,
                  float* __restrict__ out,
                  int T)
{
  __shared__ __bf16 Ash[M_BLK * A_STRIDE];           // bf16 embeddings, padded
  __shared__ float S1[M_BLK], S2[M_BLK], S3[M_BLK], Gate[M_BLK];

  const int tid  = threadIdx.x;
  const int lane = tid & 31;
  const int wave = tid >> 5;
  const long long bt0 = (long long)blockIdx.x * M_BLK;

  if (tid < M_BLK) { S1[tid] = 0.f; S2[tid] = 0.f; S3[tid] = 0.f; }

  // ---- Phase 0: hash + gather 32 tokens x 8 heads, f32 -> bf16 into LDS ----
  {
    int m = tid >> 3;            // token in block
    int h = tid & 7;             // head (0..3 -> 2gram, 4..7 -> 3gram)
    long long g = bt0 + m;
    int b = (int)(g / T);
    int t = (int)(g % T);
    const long long* row_ids = ids + (long long)b * T;
    unsigned long long id0 = (unsigned long long)row_ids[t];
    unsigned long long id1 = (t >= 1) ? (unsigned long long)row_ids[t - 1] : 0ull;
    unsigned long long id2 = (t >= 2) ? (unsigned long long)row_ids[t - 2] : 0ull;
    unsigned long long mix = id0 * (unsigned long long)mults[0]
                           ^ id1 * (unsigned long long)mults[1];
    if (h >= 4) mix ^= id2 * (unsigned long long)mults[2];
    long long mod = mods[h];
    long long r = (long long)mix % mod;       // python remainder semantics
    if (r < 0) r += mod;
    long long row = r + offs[h];
    const float* src = emb + row * PER_HEAD;
    __bf16* dst = Ash + m * A_STRIDE + h * PER_HEAD;
#pragma unroll
    for (int j = 0; j < PER_HEAD; j += 4) {
      float4 v = *(const float4*)(src + j);
      dst[j+0] = f2bf(v.x); dst[j+1] = f2bf(v.y);
      dst[j+2] = f2bf(v.z); dst[j+3] = f2bf(v.w);
    }
  }
  __syncthreads();

  // ---- Phase 1: key GEMM, reduced on the fly (key never stored) ----------
  // per-lane partials: index [s*8+r] -> token s*16 + r + (laneid>=16 ? 8 : 0)
  float s1a[16], s2a[16], s3a[16];
#pragma unroll
  for (int i = 0; i < 16; ++i) { s1a[i] = 0.f; s2a[i] = 0.f; s3a[i] = 0.f; }

  for (int tile = wave; tile < NTILES; tile += WAVES) {
    const int d0 = tile * 16;
    v8f c0 = {}; v8f c1 = {};
#pragma unroll
    for (int kk = 0; kk < KSTEPS; ++kk) {
      const int f0 = kk * 32;
      v16bf bf = load_bfrag(Wkb, d0, f0, lane);
      v16bf a0 = load_afrag(Ash, 0,  f0, lane);
      v16bf a1 = load_afrag(Ash, 16, f0, lane);
      c0 = __builtin_amdgcn_wmma_f32_16x16x32_bf16(false, a0, false, bf, (short)0, c0, false, false);
      c1 = __builtin_amdgcn_wmma_f32_16x16x32_bf16(false, a1, false, bf, (short)0, c1, false, false);
    }
    const int d = d0 + (lane & 15);
    const float wprod = kw[d] * qw[d];
    const int mofs = (lane >= 16) ? 8 : 0;
#pragma unroll
    for (int r = 0; r < 8; ++r) {
      int tok0 = r + mofs;
      float h0 = hidden[(bt0 + tok0) * HID + d];
      float k0 = c0[r];
      s1a[r] += k0 * h0 * wprod;  s2a[r] += k0 * k0;  s3a[r] += h0 * h0;
      int tok1 = 16 + r + mofs;
      float h1 = hidden[(bt0 + tok1) * HID + d];
      float k1 = c1[r];
      s1a[8+r] += k1 * h1 * wprod;  s2a[8+r] += k1 * k1;  s3a[8+r] += h1 * h1;
    }
  }

  // reduce within each 16-lane group (N direction), then combine waves in LDS
#pragma unroll
  for (int i = 0; i < 16; ++i) {
#pragma unroll
    for (int m = 1; m < 16; m <<= 1) {
      s1a[i] += __shfl_xor(s1a[i], m, 32);
      s2a[i] += __shfl_xor(s2a[i], m, 32);
      s3a[i] += __shfl_xor(s3a[i], m, 32);
    }
  }
  if ((lane & 15) == 0) {
    const int mofs = (lane >= 16) ? 8 : 0;
#pragma unroll
    for (int r = 0; r < 8; ++r) {
      atomicAdd(&S1[r + mofs],      s1a[r]);
      atomicAdd(&S2[r + mofs],      s2a[r]);
      atomicAdd(&S3[r + mofs],      s3a[r]);
      atomicAdd(&S1[16 + r + mofs], s1a[8+r]);
      atomicAdd(&S2[16 + r + mofs], s2a[8+r]);
      atomicAdd(&S3[16 + r + mofs], s3a[8+r]);
    }
  }
  __syncthreads();

  // ---- gate per token -----------------------------------------------------
  if (tid < M_BLK) {
    float rsk = rsqrtf(S2[tid] * (1.0f / HID) + RMS_EPS);
    float rsq = rsqrtf(S3[tid] * (1.0f / HID) + RMS_EPS);
    float sim = S1[tid] * rsk * rsq * (1.0f / sqrtf((float)HID));
    float v = sqrtf(fmaxf(fabsf(sim), 1e-6f));
    v = (sim > 0.f) ? v : ((sim < 0.f) ? -v : 0.f);
    Gate[tid] = 1.0f / (1.0f + __expf(-v));
  }
  __syncthreads();

  // ---- Phase 2: value GEMM, gated store ----------------------------------
  for (int tile = wave; tile < NTILES; tile += WAVES) {
    const int d0 = tile * 16;
    v8f c0 = {}; v8f c1 = {};
#pragma unroll
    for (int kk = 0; kk < KSTEPS; ++kk) {
      const int f0 = kk * 32;
      v16bf bf = load_bfrag(Wvb, d0, f0, lane);
      v16bf a0 = load_afrag(Ash, 0,  f0, lane);
      v16bf a1 = load_afrag(Ash, 16, f0, lane);
      c0 = __builtin_amdgcn_wmma_f32_16x16x32_bf16(false, a0, false, bf, (short)0, c0, false, false);
      c1 = __builtin_amdgcn_wmma_f32_16x16x32_bf16(false, a1, false, bf, (short)0, c1, false, false);
    }
    const int d = d0 + (lane & 15);
    const int mofs = (lane >= 16) ? 8 : 0;
#pragma unroll
    for (int r = 0; r < 8; ++r) {
      int tok0 = r + mofs;
      out[(bt0 + tok0) * HID + d] = Gate[tok0] * c0[r];
      int tok1 = 16 + r + mofs;
      out[(bt0 + tok1) * HID + d] = Gate[tok1] * c1[r];
    }
  }
}

extern "C" void kernel_launch(void* const* d_in, const int* in_sizes, int n_in,
                              void* d_out, int out_size, void* d_ws, size_t ws_size,
                              hipStream_t stream) {
  const float*     hidden = (const float*)d_in[0];
  const long long* ids    = (const long long*)d_in[1];   // int64 per reference
  const float*     emb    = (const float*)d_in[2];
  const float*     Wk     = (const float*)d_in[3];
  const float*     Wv     = (const float*)d_in[4];
  const float*     kw     = (const float*)d_in[5];
  const float*     qw     = (const float*)d_in[6];
  const long long* offs   = (const long long*)d_in[7];
  const long long* mults  = (const long long*)d_in[8];
  const long long* mods   = (const long long*)d_in[9];
  float* out = (float*)d_out;

  __bf16* Wkb = (__bf16*)d_ws;                 // 2 MiB
  __bf16* Wvb = Wkb + W_ELEMS;                 // 2 MiB

  // one-shot bf16 weight conversion (runs every call: deterministic)
  {
    int threads = 256, elems_per_blk = threads * 4;
    dim3 grid((W_ELEMS + elems_per_blk - 1) / elems_per_blk), block(threads);
    hipLaunchKernelGGL(convert_weights, grid, block, 0, stream, Wk, Wv, Wkb, Wvb);
  }

  const int BT = in_sizes[1];        // B*T = 16384
  const int T  = 4096;
  dim3 grid(BT / M_BLK), block(256);
  hipLaunchKernelGGL(engram_fused, grid, block, 0, stream,
                     hidden, ids, emb, Wkb, Wvb, kw, qw, offs, mults, mods, out, T);
}
